// FBSparseVGG_33809982554778
// MI455X (gfx1250) — matrix operand order repaired and checked by reading, not compiled, not run on hardware
//
#include <hip/hip_runtime.h>

typedef _Float16 f16;
typedef __attribute__((ext_vector_type(16))) _Float16     v16h;
typedef __attribute__((ext_vector_type(8)))  float        v8f;
typedef __attribute__((ext_vector_type(8)))  unsigned int v8u;
typedef __attribute__((ext_vector_type(4)))  unsigned int v4u;

union ABreg { v8u u; v16h h; };

// ---------------------------------------------------------------------------
// Fast implicit-GEMM 3x3 conv (Cin >= 16), fused BN+ReLU+mask epilogue.
// One wave computes a 16(M=pixels) x 16*NT(N=Cout) tile.
// - 9 per-tap base pointers computed once; invalid taps -> zero page, so the
//   K-loop has NO branches: A = 2x global_load_b128 (immediate offsets),
//   B = NT x 2x global_load_b128 (weights pre-packed per chunk/column),
//   then NT x v_wmma_f32_16x16x32_f16.
// Weight layout: wB[ck][n][hiHalf][j] (u32 K-pairs), ck = K/32 chunk.
// ---------------------------------------------------------------------------
template<int CIN, int NT>
__global__ void __launch_bounds__(128)
conv3x3_wmma_fast(const f16* __restrict__ in, f16* __restrict__ out,
                  const unsigned int* __restrict__ wB,
                  const float* __restrict__ scale, const float* __restrict__ bias,
                  const float* __restrict__ mask, const f16* __restrict__ zbuf,
                  int Bn, int H, int W, int Ho, int Wo, int Cout,
                  int stride, int pad)
{
    const int lane   = threadIdx.x & 31;
    const int wave   = threadIdx.x >> 5;
    const int HoWo   = Ho * Wo;
    const int M      = Bn * HoWo;
    const int mbase  = (blockIdx.x * 4 + wave) * 16;
    if (mbase >= M) return;                    // wave-uniform: EXEC stays full
    const int ncol   = blockIdx.y * (16 * NT) + (lane & 15);
    const int hiHalf = lane >> 4;              // 0/1: which K half this lane owns
    const int kOffB  = hiHalf << 3;            // 0 or 8 halves

    // This lane's A-row (output pixel)
    const int  m      = mbase + (lane & 15);
    const bool mvalid = (m < M);
    const int  mc     = mvalid ? m : 0;
    const int  b      = mc / HoWo;
    const int  rem    = mc - b * HoWo;
    const int  yo     = rem / Wo;
    const int  xo     = rem - yo * Wo;
    const int  ybase  = yo * stride - pad;
    const int  xbase  = xo * stride - pad;

    // Per-tap base pointers; out-of-image / invalid rows read the zero page.
    const f16* bases[10];
#pragma unroll
    for (int t = 0; t < 9; ++t) {
        const int ky = t / 3, kx = t - (t / 3) * 3;
        const int iy = ybase + ky, ix = xbase + kx;
        const bool ok = mvalid && (unsigned)iy < (unsigned)H
                               && (unsigned)ix < (unsigned)W;
        bases[t] = ok ? (in + (((size_t)b * H + iy) * W + ix) * CIN) : zbuf;
    }
    bases[9] = zbuf;   // padding tap (used only by the CIN==16 schedule)

    // B fragment pointer: lane's 8 u32 K-pairs are contiguous (32B) per chunk.
    const unsigned int* bp = wB + ((size_t)ncol * 16 + (hiHalf << 3));
    const size_t bstep = (size_t)Cout * 16;

    v8f acc0 = {}, acc1 = {};
    auto doChunk = [&](const f16* loBase, int loOff, const f16* hiBase, int hiOff) {
        ABreg a, b0, b1;
        const v4u alo = *(const v4u*)(loBase + loOff + kOffB);
        const v4u ahi = *(const v4u*)(hiBase + hiOff + kOffB);
#pragma unroll
        for (int j = 0; j < 4; ++j) { a.u[j] = alo[j]; a.u[j + 4] = ahi[j]; }
        b0.u = *(const v8u*)bp;
        if (NT == 2) b1.u = *(const v8u*)(bp + 256);   // next 16 columns: +1024B
        bp += bstep;
        acc0 = __builtin_amdgcn_wmma_f32_16x16x32_f16(
                   false, a.h, false, b0.h, (short)0, acc0, false, false);
        if (NT == 2)
            acc1 = __builtin_amdgcn_wmma_f32_16x16x32_f16(
                       false, a.h, false, b1.h, (short)0, acc1, false, false);
    };

    if (CIN == 16) {
        // K=144 padded to 160: 5 chunks, each spanning two taps (2ck, 2ck+1).
#pragma unroll
        for (int ck = 0; ck < 5; ++ck)
            doChunk(bases[2 * ck], 0, bases[2 * ck + 1], 0);
    } else {
        // Chunks nest inside taps: K-major == tap-major order.
#pragma unroll
        for (int t = 0; t < 9; ++t)
#pragma unroll
            for (int c0 = 0; c0 < CIN; c0 += 32)
                doChunk(bases[t], c0, bases[t], c0 + 16);
    }

    // Fused BN + ReLU + submanifold mask epilogue.
    const float s0 = scale[ncol], bb0 = bias[ncol];
    float s1 = 0.0f, bb1 = 0.0f;
    if (NT == 2) { s1 = scale[ncol + 16]; bb1 = bias[ncol + 16]; }
    const int mrow0 = mbase + (hiHalf << 3);
#pragma unroll
    for (int r = 0; r < 8; ++r) {
        const int mm = mrow0 + r;
        if (mm < M) {
            const float mk = mask[mm];
            out[(size_t)mm * Cout + ncol] =
                (f16)(fmaxf(acc0[r] * s0 + bb0, 0.0f) * mk);
            if (NT == 2)
                out[(size_t)mm * Cout + ncol + 16] =
                    (f16)(fmaxf(acc1[r] * s1 + bb1, 0.0f) * mk);
        }
    }
}

// ---------------------------------------------------------------------------
// Generic per-element gather conv: used only for layer 0 (Cin=2, K=18->32).
// ---------------------------------------------------------------------------
__global__ void __launch_bounds__(128)
conv3x3_wmma_generic(const f16* __restrict__ in, f16* __restrict__ out,
                     const unsigned int* __restrict__ wB,
                     const float* __restrict__ scale, const float* __restrict__ bias,
                     const float* __restrict__ mask,
                     int Bn, int H, int W, int Ho, int Wo,
                     int Cin, int cinLog2, int Cout, int Kreal, int Kp,
                     int stride, int pad)
{
    const int lane   = threadIdx.x & 31;
    const int wave   = threadIdx.x >> 5;
    const int HoWo   = Ho * Wo;
    const int M      = Bn * HoWo;
    const int mbase  = (blockIdx.x * 4 + wave) * 16;
    if (mbase >= M) return;
    const int ncol   = blockIdx.y * 16 + (lane & 15);
    const int hiHalf = lane >> 4;
    const int kOffB  = hiHalf << 3;

    const int  m      = mbase + (lane & 15);
    const bool mvalid = (m < M);
    const int  mc     = mvalid ? m : 0;
    const int  b      = mc / HoWo;
    const int  rem    = mc - b * HoWo;
    const int  yo     = rem / Wo;
    const int  xo     = rem - yo * Wo;
    const int  ybase  = yo * stride - pad;
    const int  xbase  = xo * stride - pad;

    const unsigned int* bp = wB + ((size_t)ncol * 16 + (hiHalf << 3));
    const size_t bstep = (size_t)Cout * 16;

    v8f acc = {};
    for (int k0 = 0; k0 < Kp; k0 += 32) {
        ABreg a, bm;
#pragma unroll
        for (int j = 0; j < 8; ++j) {
            const int k = k0 + ((j & 4) ? 16 : 0) + kOffB + ((j & 3) << 1);
            unsigned int val = 0u;
            if (mvalid && k < Kreal) {
                const int kyx = k >> cinLog2;
                const int cin = k & (Cin - 1);
                const int ky  = kyx / 3;
                const int kx  = kyx - ky * 3;
                const int iy  = ybase + ky;
                const int ix  = xbase + kx;
                if ((unsigned)iy < (unsigned)H && (unsigned)ix < (unsigned)W)
                    val = *(const unsigned int*)
                        (in + ((((size_t)b * H + iy) * W + ix) << cinLog2) + cin);
            }
            a.u[j] = val;
        }
        bm.u = *(const v8u*)bp;
        bp += bstep;
        acc = __builtin_amdgcn_wmma_f32_16x16x32_f16(
                  false, a.h, false, bm.h, (short)0, acc, false, false);
    }

    const float s  = scale[ncol];
    const float bb = bias[ncol];
    const int mrow0 = mbase + (hiHalf << 3);
#pragma unroll
    for (int r = 0; r < 8; ++r) {
        const int mm = mrow0 + r;
        if (mm < M) {
            const float v = fmaxf(acc[r] * s + bb, 0.0f) * mask[mm];
            out[(size_t)mm * Cout + ncol] = (f16)v;
        }
    }
}

// ---------------------------------------------------------------------------
// 3x3 stride-2 VALID maxpool, f16 NHWC (pooled dims always in range: H odd).
// ---------------------------------------------------------------------------
__global__ void pool3s2_f16(const f16* __restrict__ in, f16* __restrict__ out,
                            int Bn, int H, int W, int C, int Ho, int Wo)
{
    size_t idx = (size_t)blockIdx.x * blockDim.x + threadIdx.x;
    const size_t total = (size_t)Bn * Ho * Wo * C;
    if (idx >= total) return;
    const int c = (int)(idx % C);
    size_t p = idx / C;
    const int xo = (int)(p % Wo); p /= Wo;
    const int yo = (int)(p % Ho);
    const int b  = (int)(p / Ho);
    float mx = -3.0e38f;
#pragma unroll
    for (int dy = 0; dy < 3; ++dy)
#pragma unroll
        for (int dx = 0; dx < 3; ++dx) {
            const float v = (float)in[(((size_t)b * H + (yo * 2 + dy)) * W
                                       + (xo * 2 + dx)) * C + c];
            mx = fmaxf(mx, v);
        }
    out[idx] = (f16)mx;
}

__global__ void pool3s2_f32(const float* __restrict__ in, float* __restrict__ out,
                            int Bn, int H, int W, int Ho, int Wo)
{
    size_t idx = (size_t)blockIdx.x * blockDim.x + threadIdx.x;
    const size_t total = (size_t)Bn * Ho * Wo;
    if (idx >= total) return;
    const int xo = (int)(idx % Wo);
    size_t p = idx / Wo;
    const int yo = (int)(p % Ho);
    const int b  = (int)(p / Ho);
    float mx = -3.0e38f;
#pragma unroll
    for (int dy = 0; dy < 3; ++dy)
#pragma unroll
        for (int dx = 0; dx < 3; ++dx)
            mx = fmaxf(mx, in[((size_t)b * H + (yo * 2 + dy)) * W + (xo * 2 + dx)]);
    out[idx] = mx;
}

__global__ void f32_to_f16(const float* __restrict__ in, f16* __restrict__ out, size_t n)
{
    const size_t i = (size_t)blockIdx.x * blockDim.x + threadIdx.x;
    if (i < n) out[i] = (f16)in[i];
}

__global__ void zero_fill_u32(unsigned int* __restrict__ p, int n)
{
    const int i = blockIdx.x * blockDim.x + threadIdx.x;
    if (i < n) p[i] = 0u;
}

// Fold BN into per-channel scale/bias: y = x*scale + bias.
__global__ void bn_fold(const float* __restrict__ g, const float* __restrict__ be,
                        const float* __restrict__ mn, const float* __restrict__ vr,
                        float* __restrict__ scale, float* __restrict__ bias, int n)
{
    const int i = blockIdx.x * blockDim.x + threadIdx.x;
    if (i >= n) return;
    const float s = g[i] * rsqrtf(vr[i] + 1e-4f);
    scale[i] = s;
    bias[i]  = be[i] - mn[i] * s;
}

// HWIO f32 -> packed f16 B fragments: wB[ck][n][hiHalf][j] u32 holds K-pair
// (K = 32*ck + 16*(j>>2) + 8*h + 2*(j&3), K+1) for column n. Zero-padded past Kreal.
__global__ void pack_weights(const float* __restrict__ w, unsigned int* __restrict__ wp,
                             int Kreal, int Kp, int Cout)
{
    const size_t idx = (size_t)blockIdx.x * blockDim.x + threadIdx.x;
    const size_t total = (size_t)(Kp >> 5) * Cout * 16;
    if (idx >= total) return;
    size_t t = idx;
    const int j  = (int)(t & 7);  t >>= 3;
    const int h  = (int)(t & 1);  t >>= 1;
    const int n  = (int)(t % Cout);
    const int ck = (int)(t / Cout);
    const int k  = 32 * ck + 16 * (j >> 2) + 8 * h + 2 * (j & 3);
    const float lo = (k     < Kreal) ? w[(size_t)k * Cout + n]       : 0.0f;
    const float hi = (k + 1 < Kreal) ? w[(size_t)(k + 1) * Cout + n] : 0.0f;
    union { f16 hh[2]; unsigned int u; } cvt;
    cvt.hh[0] = (f16)lo;
    cvt.hh[1] = (f16)hi;
    wp[idx] = cvt.u;
}

// Head: feat = NCHW-flatten of [32,2,3,256]; out = feat @ W^T + b (tiny).
__global__ void linear_head(const f16* __restrict__ h, const float* __restrict__ Wl,
                            const float* __restrict__ bl, float* __restrict__ out)
{
    const int idx = blockIdx.x * blockDim.x + threadIdx.x;
    if (idx >= 32 * 101) return;
    const int b = idx / 101;
    const int j = idx - b * 101;
    float s = bl[j];
    for (int c = 0; c < 256; ++c)
#pragma unroll
        for (int y = 0; y < 2; ++y)
#pragma unroll
            for (int x = 0; x < 3; ++x)
                s += (float)h[(((size_t)b * 2 + y) * 3 + x) * 256 + c]
                     * Wl[(size_t)j * 1536 + c * 6 + y * 3 + x];
    out[idx] = s;
}

// ---------------------------------------------------------------------------
extern "C" void kernel_launch(void* const* d_in, const int* in_sizes, int n_in,
                              void* d_out, int out_size, void* d_ws, size_t ws_size,
                              hipStream_t stream)
{
    (void)in_sizes; (void)n_in; (void)out_size; (void)ws_size;

    const float* x     = (const float*)d_in[0];
    const float* mask0 = (const float*)d_in[1];
    const float* convW[12];
    for (int i = 0; i < 11; ++i) convW[i] = (const float*)d_in[2 + i];
    convW[11] = (const float*)d_in[13];                    // w_final
    const float *bnG[12], *bnB[12], *bnM[12], *bnV[12];
    for (int i = 0; i < 12; ++i) {
        bnG[i] = (const float*)d_in[14 + i];
        bnB[i] = (const float*)d_in[26 + i];
        bnM[i] = (const float*)d_in[38 + i];
        bnV[i] = (const float*)d_in[50 + i];
    }
    const float* Wlin = (const float*)d_in[62];
    const float* blin = (const float*)d_in[63];
    float* outp = (float*)d_out;

    // network geometry
    const int Hs[7]  = {191, 95, 47, 23, 11, 5, 2};
    const int Ws_[7] = {255, 127, 63, 31, 15, 7, 3};
    const int layerCin[12]  = {2, 16, 16, 32, 32, 64, 64, 128, 128, 256, 256, 512};
    const int layerCout[12] = {16, 16, 32, 32, 64, 64, 128, 128, 256, 256, 512, 256};
    const int layerLvl[12]  = {0, 0, 1, 1, 2, 2, 3, 3, 4, 4, 5, 5};
    const int BN_ = 32;

    // ---- workspace carve ----
    char* wsb = (char*)d_ws;
    size_t off = 0;
    auto carve = [&](size_t bytes) -> void* {
        void* p = (void*)(wsb + off);
        off = (off + bytes + 255) & ~(size_t)255;
        return p;
    };
    const size_t actElems = (size_t)BN_ * 191 * 255 * 16;   // largest activation (f16)
    f16* bufA = (f16*)carve(actElems * sizeof(f16));
    f16* bufB = (f16*)carve(actElems * sizeof(f16));
    f16* zbuf = (f16*)carve(4096);                          // zero page for OOB taps
    const float* maskLvl[7];
    float* maskBuf[7];
    maskLvl[0] = mask0;
    for (int l = 1; l <= 6; ++l) {
        maskBuf[l] = (float*)carve((size_t)BN_ * Hs[l] * Ws_[l] * sizeof(float));
        maskLvl[l] = maskBuf[l];
    }
    float *bnScale[12], *bnBias[12];
    for (int i = 0; i < 12; ++i) {
        bnScale[i] = (float*)carve((size_t)layerCout[i] * sizeof(float));
        bnBias[i]  = (float*)carve((size_t)layerCout[i] * sizeof(float));
    }
    unsigned int* wpk[12];
    int Kreal[12], Kp[12];
    for (int i = 0; i < 12; ++i) {
        Kreal[i] = 9 * layerCin[i];
        Kp[i]    = (Kreal[i] + 31) & ~31;
        wpk[i]   = (unsigned int*)carve((size_t)(Kp[i] / 2) * layerCout[i]
                                        * sizeof(unsigned int));
    }

    // ---- 0) zero page ----
    zero_fill_u32<<<4, 256, 0, stream>>>((unsigned int*)zbuf, 1024);

    // ---- 1) fold BN, pack weights ----
    for (int i = 0; i < 12; ++i) {
        const int d = layerCout[i];
        bn_fold<<<(d + 255) / 256, 256, 0, stream>>>(bnG[i], bnB[i], bnM[i], bnV[i],
                                                     bnScale[i], bnBias[i], d);
        const size_t nel = (size_t)(Kp[i] >> 5) * layerCout[i] * 16;
        pack_weights<<<(unsigned)((nel + 255) / 256), 256, 0, stream>>>(
            convW[i], wpk[i], Kreal[i], Kp[i], layerCout[i]);
    }

    // ---- 2) pooled activity masks per resolution ----
    for (int l = 1; l <= 6; ++l) {
        const size_t nel = (size_t)BN_ * Hs[l] * Ws_[l];
        pool3s2_f32<<<(unsigned)((nel + 255) / 256), 256, 0, stream>>>(
            maskLvl[l - 1], maskBuf[l], BN_, Hs[l - 1], Ws_[l - 1], Hs[l], Ws_[l]);
    }

    // ---- 3) input f32 -> f16 (already masked) ----
    {
        const size_t nel = (size_t)BN_ * 191 * 255 * 2;
        f32_to_f16<<<(unsigned)((nel + 255) / 256), 256, 0, stream>>>(x, bufA, nel);
    }

    // ---- 4) conv stack with ping-pong buffers ----
    f16* cur = bufA;
    f16* nxt = bufB;
    auto runConv = [&](int i, int Hin, int Win, int Hout, int Wout,
                       int stride, int pad, const float* mk) {
        const int Cin = layerCin[i], Cout = layerCout[i];
        const int M = BN_ * Hout * Wout;
        const int mtiles = (M + 15) / 16;
        if (Cin == 2) {
            dim3 grid((unsigned)((mtiles + 3) / 4), (unsigned)(Cout / 16));
            conv3x3_wmma_generic<<<grid, 128, 0, stream>>>(
                cur, nxt, wpk[i], bnScale[i], bnBias[i], mk,
                BN_, Hin, Win, Hout, Wout, Cin, 1, Cout, Kreal[i], Kp[i],
                stride, pad);
        } else {
            const int NT = (Cout >= 32) ? 2 : 1;
            dim3 grid((unsigned)((mtiles + 3) / 4), (unsigned)(Cout / (16 * NT)));
#define LAUNCH_FAST(CC, NN)                                                     \
            conv3x3_wmma_fast<CC, NN><<<grid, 128, 0, stream>>>(                \
                cur, nxt, wpk[i], bnScale[i], bnBias[i], mk, zbuf,              \
                BN_, Hin, Win, Hout, Wout, Cout, stride, pad)
            if      (Cin == 16 && NT == 1) LAUNCH_FAST(16, 1);
            else if (Cin == 16)            LAUNCH_FAST(16, 2);
            else if (Cin == 32)            LAUNCH_FAST(32, 2);
            else if (Cin == 64)            LAUNCH_FAST(64, 2);
            else if (Cin == 128)           LAUNCH_FAST(128, 2);
            else if (Cin == 256)           LAUNCH_FAST(256, 2);
            else                           LAUNCH_FAST(512, 2);
#undef LAUNCH_FAST
        }
        f16* t = cur; cur = nxt; nxt = t;
    };
    auto runPool = [&](int lvlIn, int C) {
        const int Hin = Hs[lvlIn], Win = Ws_[lvlIn];
        const int Hout = Hs[lvlIn + 1], Wout = Ws_[lvlIn + 1];
        const size_t nel = (size_t)BN_ * Hout * Wout * C;
        pool3s2_f16<<<(unsigned)((nel + 255) / 256), 256, 0, stream>>>(
            cur, nxt, BN_, Hin, Win, C, Hout, Wout);
        f16* t = cur; cur = nxt; nxt = t;
    };

    for (int i = 0; i < 11; ++i) {
        const int lvl = layerLvl[i];
        runConv(i, Hs[lvl], Ws_[lvl], Hs[lvl], Ws_[lvl], 1, 1, maskLvl[lvl]);
        if (i == 1 || i == 3 || i == 5 || i == 7 || i == 9)
            runPool(lvl, layerCout[i]);
    }
    // final strided sparse conv 512->256, k3 s2 VALID: 5x7 -> 2x3, masked by m6
    runConv(11, Hs[5], Ws_[5], Hs[6], Ws_[6], 2, 0, maskLvl[6]);

    // ---- 5) linear head ----
    linear_head<<<(32 * 101 + 255) / 256, 256, 0, stream>>>(cur, Wlin, blin, outp);
}
